// myBlock_10307921511065
// MI455X (gfx1250) — compile-verified, matrix operand order checked by reading, not run on hardware
//
#include <hip/hip_runtime.h>
#include <hip/hip_bf16.h>

typedef unsigned short u16;
typedef unsigned int   u32;
typedef __attribute__((ext_vector_type(2)))  float  v2f;
typedef __attribute__((ext_vector_type(8)))  float  v8f;
typedef __attribute__((ext_vector_type(16))) __bf16 v16bf;

union Frag16 {
  v16bf bf;
  uint4 u4[2];
  u32   u[8];
  u16   us[16];
};

__device__ __forceinline__ u16 f32_bf16(float f) {
  u32 x = __float_as_uint(f);
  x += 0x7FFFu + ((x >> 16) & 1u);
  return (u16)(x >> 16);
}

__device__ __forceinline__ float blockReduce256(float v, float* sh) {
  int t = threadIdx.x;
  sh[t] = v;
  __syncthreads();
#pragma unroll
  for (int off = 128; off > 0; off >>= 1) {
    if (t < off) sh[t] += sh[t + off];
    __syncthreads();
  }
  float r = sh[0];
  __syncthreads();
  return r;
}

// ---------------------------------------------------------------------------
// Weight repack: (co,ci,ky,kx) fp32 -> bf16 [co][k], k = (ky*3+kx)*64 + ci
// ---------------------------------------------------------------------------
__global__ __launch_bounds__(256) void wcvt(const float* __restrict__ w,
                                            u16* __restrict__ wb) {
  int id = blockIdx.x * 256 + threadIdx.x;
  if (id >= 36864) return;
  int co = id / 576;
  int k  = id - co * 576;
  int r  = k >> 6;
  int ci = k & 63;
  wb[id] = f32_bf16(w[(co * 64 + ci) * 9 + r]);
}

// ---------------------------------------------------------------------------
// 3x3 conv, 64->64 ch, pad 1, bias. Implicit GEMM via v_wmma_f32_16x16x32_bf16
// Block: 256 thr = 8 waves; covers all 64 co x 32 x-positions of one row.
// ---------------------------------------------------------------------------
__global__ __launch_bounds__(256) void conv3x3_wmma(
    const float* __restrict__ in, const u16* __restrict__ wb,
    const float* __restrict__ bias, float* __restrict__ out) {
  __shared__ u16 lds[3 * 34 * 66];  // [dy][cx][ci], padded stride 66
  const int tid = threadIdx.x;
  const int x0 = blockIdx.x * 32;
  const int y = blockIdx.y;
  const int b = blockIdx.z;

  for (int idx = tid; idx < 3 * 64 * 34; idx += 256) {
    int dy = idx / (64 * 34);
    int rem = idx - dy * (64 * 34);
    int ci = rem / 34;
    int cx = rem - ci * 34;
    int yg = y + dy - 1;
    int xg = x0 + cx - 1;
    float v = 0.f;
    if ((u32)yg < 128u && (u32)xg < 128u)
      v = in[((b * 64 + ci) * 128 + yg) * 128 + xg];
    lds[(dy * 34 + cx) * 66 + ci] = f32_bf16(v);
  }
  __syncthreads();

  const int wave = tid >> 5;
  const int lane = tid & 31;
  const int i16 = lane & 15;
  const int hi = lane >> 4;
  const int kh = hi * 8;
  const int coT = (wave >> 1) * 16;
  const int xT = (wave & 1) * 16;

  const u16* wrow = wb + (coT + i16) * 576;
  __builtin_prefetch(wrow, 0, 0);

  v8f acc = {};
#pragma unroll
  for (int r = 0; r < 9; ++r) {
    const int ky = r / 3;
    const int kx = r - ky * 3;
    const u16* lrow = &lds[(ky * 34 + (xT + i16 + kx)) * 66];
#pragma unroll
    for (int h = 0; h < 2; ++h) {
      const int cb = h * 32;
      Frag16 af, bq;
      const u16* wp = wrow + r * 64 + cb + kh;  // 16B aligned
      af.u4[0] = *(const uint4*)(wp);
      af.u4[1] = *(const uint4*)(wp + 16);
      const u16* lp = lrow + cb + kh;
#pragma unroll
      for (int v = 0; v < 4; ++v) {
        bq.u[v]     = *(const u32*)(lp + 2 * v);
        bq.u[4 + v] = *(const u32*)(lp + 16 + 2 * v);
      }
      acc = __builtin_amdgcn_wmma_f32_16x16x32_bf16(
          false, af.bf, false, bq.bf, (short)0, acc, false, false);
    }
  }

  const int xo = x0 + xT + i16;
#pragma unroll
  for (int v = 0; v < 8; ++v) {
    int co = coT + v + 8 * hi;
    out[((b * 64 + co) * 128 + y) * 128 + xo] = acc[v] + bias[co];
  }
}

// ---------------------------------------------------------------------------
// Batched 128x128 GEMM (256 batches) via v_wmma_f32_16x16x4_f32.
// TRANSB=true : D = A * B^T   (att = q @ k^T)
// TRANSB=false: D = A * B     (o = v @ softmax)
// grid = (8 row-tiles, 256 matrices), 8 waves = 8 col-tiles.
// ---------------------------------------------------------------------------
template <bool TRANSB>
__global__ __launch_bounds__(256) void gemm128_f32(const float* __restrict__ A,
                                                   const float* __restrict__ B,
                                                   float* __restrict__ D) {
  const int lane = threadIdx.x & 31;
  const int wave = threadIdx.x >> 5;
  const int i16 = lane & 15;
  const int hi = lane >> 4;
  const int k0 = hi * 2;
  const int mt = blockIdx.x;
  const int nt = wave;
  const int mat = blockIdx.y;

  const float* Am = A + mat * 16384 + (mt * 16 + i16) * 128;
  const float* Bm = B + mat * 16384;

  v8f acc = {};
#pragma unroll 4
  for (int k = 0; k < 128; k += 4) {
    v2f a, bb;
    a[0] = Am[k + k0];
    a[1] = Am[k + k0 + 1];
    if (TRANSB) {
      const float* bp = Bm + (nt * 16 + i16) * 128 + k + k0;
      bb[0] = bp[0];
      bb[1] = bp[1];
    } else {
      const float* bp = Bm + (k + k0) * 128 + nt * 16 + i16;
      bb[0] = bp[0];
      bb[1] = bp[128];
    }
    acc = __builtin_amdgcn_wmma_f32_16x16x4_f32(
        false, a, false, bb, (short)0, acc, false, false);
  }

  float* Dp = D + mat * 16384 + nt * 16 + i16;
#pragma unroll
  for (int v = 0; v < 8; ++v) Dp[(mt * 16 + v + 8 * hi) * 128] = acc[v];
}

// ---------------------------------------------------------------------------
// 7x7 offset conv: 64 -> 18 channels, pad 3. One thread per output element.
// ---------------------------------------------------------------------------
__global__ __launch_bounds__(256) void conv7x7_off(
    const float* __restrict__ in, const float* __restrict__ w,
    const float* __restrict__ bias, float* __restrict__ off) {
  int id = blockIdx.x * 256 + threadIdx.x;  // 4*18*128*128
  int x = id & 127;
  int y = (id >> 7) & 127;
  int t = id >> 14;
  int co = t % 18;
  int b = t / 18;
  float acc = bias[co];
  for (int ci = 0; ci < 64; ++ci) {
    const float* ip = in + (size_t)(b * 64 + ci) * 16384;
    const float* wp = w + (size_t)(co * 64 + ci) * 49;
#pragma unroll
    for (int ky = 0; ky < 7; ++ky) {
      int yy = y + ky - 3;
      if ((u32)yy >= 128u) continue;
#pragma unroll
      for (int kx = 0; kx < 7; ++kx) {
        int xx = x + kx - 3;
        if ((u32)xx >= 128u) continue;
        acc += ip[yy * 128 + xx] * wp[ky * 7 + kx];
      }
    }
  }
  off[id] = acc;
}

// ---------------------------------------------------------------------------
// Deformable: bilinear sample 9 taps + grouped 3x3 conv (G=8) + bias.
// One thread per (b, group, y, x) -> 8 output channels.
// ---------------------------------------------------------------------------
__global__ __launch_bounds__(256) void deform_gconv(
    const float* __restrict__ in, const float* __restrict__ off,
    const float* __restrict__ w, const float* __restrict__ bias,
    float* __restrict__ out) {
  int id = blockIdx.x * 256 + threadIdx.x;  // 4*8*128*128
  int x = id & 127;
  int y = (id >> 7) & 127;
  int t = id >> 14;
  int g = t & 7;
  int b = t >> 3;

  float acc[8];
#pragma unroll
  for (int o = 0; o < 8; ++o) acc[o] = 0.f;

  const float* ob = off + (size_t)b * 18 * 16384 + y * 128 + x;
#pragma unroll
  for (int tap = 0; tap < 9; ++tap) {
    float oy = ob[tap * 16384];
    float ox = ob[(9 + tap) * 16384];
    int ky = tap / 3 - 1;
    int kx = tap - (tap / 3) * 3 - 1;
    float py = fminf(fmaxf((float)(y + ky) + oy, 0.f), 127.f);
    float px = fminf(fmaxf((float)(x + kx) + ox, 0.f), 127.f);
    float y0f = floorf(py), x0f = floorf(px);
    float wy = py - y0f, wx = px - x0f;
    int y0 = (int)y0f, x0 = (int)x0f;
    int y1 = y0 + 1 > 127 ? 127 : y0 + 1;
    int x1 = x0 + 1 > 127 ? 127 : x0 + 1;
    float w00 = (1.f - wy) * (1.f - wx), w01 = (1.f - wy) * wx;
    float w10 = wy * (1.f - wx), w11 = wy * wx;
#pragma unroll
    for (int cig = 0; cig < 8; ++cig) {
      const float* ip = in + (size_t)(b * 64 + g * 8 + cig) * 16384;
      float s = w00 * ip[y0 * 128 + x0] + w01 * ip[y0 * 128 + x1] +
                w10 * ip[y1 * 128 + x0] + w11 * ip[y1 * 128 + x1];
      const float* wp = w + (g * 8) * 72 + cig * 9 + tap;
#pragma unroll
      for (int o = 0; o < 8; ++o) acc[o] += s * wp[o * 72];
    }
  }
  float* op = out + (size_t)(b * 64 + g * 8) * 16384 + y * 128 + x;
#pragma unroll
  for (int o = 0; o < 8; ++o) op[o * 16384] = acc[o] + bias[g * 8 + o];
}

// ---------------------------------------------------------------------------
// Deterministic global sum / sumsq over 4194304 floats (stage 1: 4096 blocks)
// ---------------------------------------------------------------------------
__global__ __launch_bounds__(256) void red1(const float* __restrict__ x,
                                            float* __restrict__ ps,
                                            float* __restrict__ pq) {
  __shared__ float sh[256];
  int base = blockIdx.x * 1024;
  float s = 0.f, sq = 0.f;
#pragma unroll
  for (int e = 0; e < 4; ++e) {
    float v = x[base + e * 256 + threadIdx.x];
    s += v;
    sq += v * v;
  }
  float ts = blockReduce256(s, sh);
  float tq = blockReduce256(sq, sh);
  if (threadIdx.x == 0) {
    ps[blockIdx.x] = ts;
    pq[blockIdx.x] = tq;
  }
}

__global__ __launch_bounds__(256) void red2(const float* __restrict__ ps,
                                            const float* __restrict__ pq,
                                            float* __restrict__ scal) {
  __shared__ float sh[256];
  float s = 0.f, sq = 0.f;
  for (int i = threadIdx.x; i < 4096; i += 256) {
    s += ps[i];
    sq += pq[i];
  }
  float ts = blockReduce256(s, sh);
  float tq = blockReduce256(sq, sh);
  if (threadIdx.x == 0) {
    const float n = 4194304.f;
    float var = (tq - ts * ts / n) / (n - 1.f);  // ddof=1
    scal[0] = rsqrtf(fmaxf(var, 1e-30f));        // 1/std
  }
}

// ---------------------------------------------------------------------------
// att *= 1/std; softmax over channel axis (64, stride HW). Thread per (b,i,j).
// ---------------------------------------------------------------------------
__global__ __launch_bounds__(256) void softmax_ch(float* __restrict__ att,
                                                  const float* __restrict__ scal) {
  int id = blockIdx.x * 256 + threadIdx.x;  // 65536
  int b = id >> 14;
  int pos = id & 16383;
  float* p = att + (size_t)b * 64 * 16384 + pos;
  float s = scal[0];
  float v[64];
  float mx = -1e30f;
#pragma unroll
  for (int c = 0; c < 64; ++c) {
    v[c] = p[c * 16384] * s;
    mx = fmaxf(mx, v[c]);
  }
  float sum = 0.f;
#pragma unroll
  for (int c = 0; c < 64; ++c) {
    v[c] = __expf(v[c] - mx);
    sum += v[c];
  }
  float inv = 1.f / sum;
#pragma unroll
  for (int c = 0; c < 64; ++c) p[c * 16384] = v[c] * inv;
}

// ---------------------------------------------------------------------------
// SE block
// ---------------------------------------------------------------------------
__global__ __launch_bounds__(256) void se_pool(const float* __restrict__ o,
                                               float* __restrict__ pooled) {
  __shared__ float sh[256];
  const float* p = o + (size_t)blockIdx.x * 16384;
  float s = 0.f;
  for (int i = threadIdx.x; i < 16384; i += 256) s += p[i];
  float t = blockReduce256(s, sh);
  if (threadIdx.x == 0) pooled[blockIdx.x] = t * (1.f / 16384.f);
}

__global__ __launch_bounds__(256) void se_fc(
    const float* __restrict__ pooled, const float* __restrict__ w1,
    const float* __restrict__ b1, const float* __restrict__ w2,
    const float* __restrict__ b2, float* __restrict__ scales) {
  int t = threadIdx.x;
  int b = t >> 6;
  int c2 = t & 63;
  float h[4];
#pragma unroll
  for (int j = 0; j < 4; ++j) {
    float a = b1[j];
    for (int c = 0; c < 64; ++c) a += pooled[b * 64 + c] * w1[c * 4 + j];
    h[j] = fmaxf(a, 0.f);
  }
  float sc = b2[c2];
#pragma unroll
  for (int j = 0; j < 4; ++j) sc += h[j] * w2[j * 64 + c2];
  scales[t] = 1.f / (1.f + __expf(-sc));
}

__global__ __launch_bounds__(256) void se_apply(float* __restrict__ o,
                                                const float* __restrict__ x,
                                                const float* __restrict__ scales) {
  int base = (blockIdx.x * 256 + threadIdx.x) * 4;
#pragma unroll
  for (int e = 0; e < 4; ++e) {
    int i = base + e;
    int bc = i >> 14;
    o[i] = o[i] * scales[bc] + x[i];
  }
}

// ---------------------------------------------------------------------------
// BatchNorm over (B,H,W) per channel + ReLU
// ---------------------------------------------------------------------------
__global__ __launch_bounds__(256) void bn_stats1(const float* __restrict__ t,
                                                 float* __restrict__ ps,
                                                 float* __restrict__ pq) {
  __shared__ float sh[256];
  int c = blockIdx.x >> 4;
  int chunk = blockIdx.x & 15;
  int b = chunk >> 2;
  int q = chunk & 3;
  const float* p = t + (size_t)(b * 64 + c) * 16384 + q * 4096;
  float s = 0.f, sq = 0.f;
  for (int i = threadIdx.x; i < 4096; i += 256) {
    float v = p[i];
    s += v;
    sq += v * v;
  }
  float ts = blockReduce256(s, sh);
  float tq = blockReduce256(sq, sh);
  if (threadIdx.x == 0) {
    ps[blockIdx.x] = ts;
    pq[blockIdx.x] = tq;
  }
}

__global__ void bn_stats2(const float* __restrict__ ps,
                          const float* __restrict__ pq,
                          float* __restrict__ mu, float* __restrict__ iv) {
  int c = threadIdx.x;
  if (c < 64) {
    float s = 0.f, sq = 0.f;
    for (int i = 0; i < 16; ++i) {
      s += ps[c * 16 + i];
      sq += pq[c * 16 + i];
    }
    float m = s * (1.f / 262144.f);
    float var = sq * (1.f / 262144.f) - m * m;  // biased (ddof=0)
    mu[c] = m;
    iv[c] = rsqrtf(var + 1e-5f);
  }
}

__global__ __launch_bounds__(256) void bn_apply(
    const float* __restrict__ t, const float* __restrict__ g,
    const float* __restrict__ be, const float* __restrict__ mu,
    const float* __restrict__ iv, float* __restrict__ out) {
  int base = (blockIdx.x * 256 + threadIdx.x) * 4;
#pragma unroll
  for (int e = 0; e < 4; ++e) {
    int i = base + e;
    int c = (i >> 14) & 63;
    out[i] = fmaxf(g[c] * (t[i] - mu[c]) * iv[c] + be[c], 0.f);
  }
}

// ---------------------------------------------------------------------------
extern "C" void kernel_launch(void* const* d_in, const int* in_sizes, int n_in,
                              void* d_out, int out_size, void* d_ws,
                              size_t ws_size, hipStream_t stream) {
  (void)in_sizes; (void)n_in; (void)out_size; (void)ws_size;
  const float* x1 = (const float*)d_in[0];
  const float* x2 = (const float*)d_in[1];
  const float* cw[6] = {(const float*)d_in[2],  (const float*)d_in[4],
                        (const float*)d_in[6],  (const float*)d_in[8],
                        (const float*)d_in[10], (const float*)d_in[12]};
  const float* cb[6] = {(const float*)d_in[3],  (const float*)d_in[5],
                        (const float*)d_in[7],  (const float*)d_in[9],
                        (const float*)d_in[11], (const float*)d_in[13]};
  const float* br1ow = (const float*)d_in[14];
  const float* br1ob = (const float*)d_in[15];
  const float* br1w  = (const float*)d_in[16];
  const float* br1b  = (const float*)d_in[17];
  const float* br2ow = (const float*)d_in[18];
  const float* br2ob = (const float*)d_in[19];
  const float* br2w  = (const float*)d_in[20];
  const float* br2b  = (const float*)d_in[21];
  const float* sew1  = (const float*)d_in[22];
  const float* seb1  = (const float*)d_in[23];
  const float* sew2  = (const float*)d_in[24];
  const float* seb2  = (const float*)d_in[25];
  const float* ab3w  = (const float*)d_in[26];
  const float* ab3b  = (const float*)d_in[27];
  const float* ab3g  = (const float*)d_in[28];
  const float* ab3be = (const float*)d_in[29];
  const float* ab4w  = (const float*)d_in[30];
  const float* ab4b  = (const float*)d_in[31];
  const float* ab4g  = (const float*)d_in[32];
  const float* ab4be = (const float*)d_in[33];
  float* out = (float*)d_out;

  const size_t SZ = 4194304, OFFSZ = 1179648;
  float* wsf = (float*)d_ws;
  float* B[9];
  for (int i = 0; i < 9; ++i) B[i] = wsf + (size_t)i * SZ;
  float* off1 = wsf + 9 * SZ;
  float* off2 = off1 + OFFSZ;
  u16* wbz = (u16*)(wsf + 9 * SZ + 2 * OFFSZ);  // 8 * 36864 bf16
  float* sm = wsf + 9 * SZ + 2 * OFFSZ + 147456;
  float* psum   = sm;
  float* pq     = sm + 4096;
  float* scal   = sm + 8192;
  float* pooled = sm + 8200;
  float* scales = sm + 8456;
  float* bnps   = sm + 8712;
  float* bnpq   = sm + 9736;
  float* bnmu   = sm + 10760;
  float* bniv   = sm + 10824;

  const float* allw[8] = {cw[0], cw[1], cw[2], cw[3], cw[4], cw[5], ab3w, ab4w};
  for (int i = 0; i < 8; ++i)
    wcvt<<<144, 256, 0, stream>>>(allw[i], wbz + (size_t)i * 36864);

  dim3 cg(4, 128, 4);
  conv3x3_wmma<<<cg, 256, 0, stream>>>(x1, wbz + 0 * 36864, cb[0], B[0]);  // q1
  conv3x3_wmma<<<cg, 256, 0, stream>>>(x2, wbz + 1 * 36864, cb[1], B[1]);  // q2
  conv3x3_wmma<<<cg, 256, 0, stream>>>(x1, wbz + 2 * 36864, cb[2], B[2]);  // kc1
  conv3x3_wmma<<<cg, 256, 0, stream>>>(x2, wbz + 3 * 36864, cb[3], B[3]);  // kc2
  conv3x3_wmma<<<cg, 256, 0, stream>>>(x1, wbz + 4 * 36864, cb[4], B[4]);  // v1
  conv3x3_wmma<<<cg, 256, 0, stream>>>(x2, wbz + 5 * 36864, cb[5], B[5]);  // v2

  conv7x7_off<<<4608, 256, 0, stream>>>(B[2], br1ow, br1ob, off1);
  conv7x7_off<<<4608, 256, 0, stream>>>(B[3], br2ow, br2ob, off2);
  deform_gconv<<<2048, 256, 0, stream>>>(B[2], off1, br1w, br1b, B[6]);  // kd1
  deform_gconv<<<2048, 256, 0, stream>>>(B[3], off2, br2w, br2b, B[7]);  // kd2

  dim3 gg(8, 256);
  // branch 1: att1 = q1 @ kd2^T ; /std ; softmax(ch) ; o1 = v1 @ s1
  gemm128_f32<true><<<gg, 256, 0, stream>>>(B[0], B[7], B[8]);
  red1<<<4096, 256, 0, stream>>>(B[8], psum, pq);
  red2<<<1, 256, 0, stream>>>(psum, pq, scal);
  softmax_ch<<<256, 256, 0, stream>>>(B[8], scal);
  gemm128_f32<false><<<gg, 256, 0, stream>>>(B[4], B[8], B[2]);  // o1 -> B2
  // branch 2
  gemm128_f32<true><<<gg, 256, 0, stream>>>(B[1], B[6], B[8]);
  red1<<<4096, 256, 0, stream>>>(B[8], psum, pq);
  red2<<<1, 256, 0, stream>>>(psum, pq, scal);
  softmax_ch<<<256, 256, 0, stream>>>(B[8], scal);
  gemm128_f32<false><<<gg, 256, 0, stream>>>(B[5], B[8], B[3]);  // o2 -> B3

  // SE + residual
  se_pool<<<256, 256, 0, stream>>>(B[2], pooled);
  se_fc<<<1, 256, 0, stream>>>(pooled, sew1, seb1, sew2, seb2, scales);
  se_apply<<<4096, 256, 0, stream>>>(B[2], x1, scales);
  se_pool<<<256, 256, 0, stream>>>(B[3], pooled);
  se_fc<<<1, 256, 0, stream>>>(pooled, sew1, seb1, sew2, seb2, scales);
  se_apply<<<4096, 256, 0, stream>>>(B[3], x2, scales);

  // final conv + BN + ReLU
  conv3x3_wmma<<<cg, 256, 0, stream>>>(B[2], wbz + 6 * 36864, ab3b, B[0]);
  conv3x3_wmma<<<cg, 256, 0, stream>>>(B[3], wbz + 7 * 36864, ab4b, B[1]);
  bn_stats1<<<1024, 256, 0, stream>>>(B[0], bnps, bnpq);
  bn_stats2<<<1, 64, 0, stream>>>(bnps, bnpq, bnmu, bniv);
  bn_apply<<<4096, 256, 0, stream>>>(B[0], ab3g, ab3be, bnmu, bniv, out);
  bn_stats1<<<1024, 256, 0, stream>>>(B[1], bnps, bnpq);
  bn_stats2<<<1, 64, 0, stream>>>(bnps, bnpq, bnmu, bniv);
  bn_apply<<<4096, 256, 0, stream>>>(B[1], ab4g, ab4be, bnmu, bniv, out + SZ);
}
